// EpiNN_att_40278203302207
// MI455X (gfx1250) — matrix operand order repaired and checked by reading, not compiled, not run on hardware
//
#include <hip/hip_runtime.h>

// ---------------------------------------------------------------------------
// Problem constants
// ---------------------------------------------------------------------------
#define NB   64
#define LL   512
#define DD   1280
#define DH   32
#define KT   (DD / 32)                 // 40 K-chunks of 32
#define ROWS 128                       // L-rows per proj block
#define TRI  ((LL/16) * (LL/16 + 1) / 2)   // 528 upper-tri tile pairs
#define JOBS (NB * TRI)                    // 33792 wave-jobs
#define SCALE_ATTN 0.08838834764831845f    // 1/(N_HEADS*sqrt(HEAD_DIM))

typedef __attribute__((ext_vector_type(16))) __bf16 v16bf;
typedef __attribute__((ext_vector_type(8)))  float  v8f;

union Frag16 {
    unsigned short u[16];
    unsigned       p32[8];
    uint4          q[2];
    v16bf          v;
};

// pack two f32 -> bf16x2. Prefer HW packed convert; else round-half-up
// (v_add_nc_u32 x2) + one v_perm_b32 grabbing both high halves: 3 VALU / 2 elem.
#if defined(__has_builtin)
#if __has_builtin(__builtin_amdgcn_cvt_pk_bf16_f32)
#define HAVE_PK_BF16 1
#endif
#endif

__device__ __forceinline__ unsigned pack_bf2(float a, float b) {
#ifdef HAVE_PK_BF16
    typedef __attribute__((ext_vector_type(2))) __bf16 v2bf;
    v2bf r = __builtin_amdgcn_cvt_pk_bf16_f32(a, b);
    return __builtin_bit_cast(unsigned, r);
#else
    unsigned ua = __builtin_bit_cast(unsigned, a) + 0x8000u;
    unsigned ub = __builtin_bit_cast(unsigned, b) + 0x8000u;
    // result = { ub[31:16], ua[31:16] }
    return __builtin_amdgcn_perm(ub, ua, 0x07060302u);
#endif
}

__device__ __forceinline__ unsigned short f2bf1(float f) {
    return (unsigned short)((__builtin_bit_cast(unsigned, f) + 0x8000u) >> 16);
}

// CDNA5 async global->LDS copy (ASYNCcnt-tracked, CDNA5 ISA)
__device__ __forceinline__ void async_ld_b128(void* lds, const void* g) {
    asm volatile("global_load_async_to_lds_b128 %0, %1, off"
                 :: "v"((unsigned)(size_t)lds),
                    "v"((unsigned long long)(size_t)g)
                 : "memory");
}
#define WAIT_ASYNC(n) asm volatile("s_wait_asynccnt %0" :: "n"(n) : "memory")

// ---------------------------------------------------------------------------
// Kernel 0a: out[n] = b_seq
// ---------------------------------------------------------------------------
__global__ void init_out_kernel(float* __restrict__ out,
                                const float* __restrict__ b_seq) {
    int n = threadIdx.x;
    if (n < NB) out[n] = b_seq[0];
}

// ---------------------------------------------------------------------------
// Kernel 0b: W_proj (32x1280 f32) -> bf16 row-major (80 KB, L2-resident).
// ---------------------------------------------------------------------------
__global__ void wconv_kernel(const float* __restrict__ W,
                             unsigned* __restrict__ Wb32) {
    int i = blockIdx.x * 256 + threadIdx.x;   // 80 blocks x 256 = 20480 pairs
    Wb32[i] = pack_bf2(W[2 * i], W[2 * i + 1]);
}

// ---------------------------------------------------------------------------
// Kernel 1: H = x @ W_proj^T (bf16 WMMA, f32 accum) + fused first_order.
// grid: NB * (LL/ROWS) = 256 blocks, 256 threads (8 waves).
// x chunk (128 rows x 32 k fp32) double-buffered into LDS via async copies.
// Each wave owns a 16-row strip, computes both 16-col tiles from one A frag;
// the 8 A-fragments tile the chunk exactly once, so the w_token dot is fused
// into the A build (no second LDS pass).
// ---------------------------------------------------------------------------
__global__ void proj_kernel(const float* __restrict__ x,
                            const float* __restrict__ w_token,
                            const float* __restrict__ w_seq,
                            const unsigned short* __restrict__ Wb,
                            unsigned short* __restrict__ H,   // [NB][LL][DH] bf16
                            float* __restrict__ out) {
    __shared__ float xb[2][ROWS * DH];   // 2 x 16 KB fp32 staging
    __shared__ float red[256];

    const int tid  = threadIdx.x;
    const int lane = tid & 31;
    const int wid  = tid >> 5;
    const int n    = blockIdx.x >> 2;          // 4 strips of 128 rows per n
    const int l0   = (blockIdx.x & 3) * ROWS;

    const float* xbase = x + ((size_t)n * LL + l0) * DD;

    // 1024 b128 copies per chunk / 256 threads = 4 each
    auto issue = [&](int kc, int buf) {
        #pragma unroll
        for (int i = 0; i < 4; ++i) {
            const int flat = tid + i * 256;
            const int row  = flat >> 3;
            const int seg  = flat & 7;
            async_ld_b128(&xb[buf][row * DH + seg * 4],
                          xbase + (size_t)row * DD + kc * 32 + seg * 4);
        }
    };

    // per-lane coords (ISA fragment layouts)
    const int rl = wid * 16 + (lane & 15);      // A row in strip
    const int kb = (lane < 16) ? 0 : 8;         // A: K kb..kb+7 and 16+kb..
    const int j0 = lane & 15;                   // B col (+16 for second tile)
    const int k0 = (lane < 16) ? 0 : 8;         // B: K k0..k0+7 and 16+k0..

    v8f acc0 = {}, acc1 = {};
    float s_fo = 0.0f;   // partial of x[rl] . w_token over this lane's K slice

    issue(0, 0);
    for (int kc = 0; kc < KT; ++kc) {
        if (kc + 1 < KT) { issue(kc + 1, (kc + 1) & 1); WAIT_ASYNC(4); }
        else             { WAIT_ASYNC(0); }
        __syncthreads();

        const float* buf = xb[kc & 1];

        // ---- A fragment build (4x ds b128) + fused token dot
        Frag16 A;
        {
            const float* p = buf + rl * DH;
            float4 a0 = *(const float4*)(p + kb);
            float4 a1 = *(const float4*)(p + kb + 4);
            float4 a2 = *(const float4*)(p + 16 + kb);
            float4 a3 = *(const float4*)(p + 16 + kb + 4);

            const float* wt = w_token + kc * 32;
            float4 t0 = *(const float4*)(wt + kb);
            float4 t1 = *(const float4*)(wt + kb + 4);
            float4 t2 = *(const float4*)(wt + 16 + kb);
            float4 t3 = *(const float4*)(wt + 16 + kb + 4);
            s_fo += a0.x*t0.x + a0.y*t0.y + a0.z*t0.z + a0.w*t0.w
                  + a1.x*t1.x + a1.y*t1.y + a1.z*t1.z + a1.w*t1.w
                  + a2.x*t2.x + a2.y*t2.y + a2.z*t2.z + a2.w*t2.w
                  + a3.x*t3.x + a3.y*t3.y + a3.z*t3.z + a3.w*t3.w;

            A.p32[0] = pack_bf2(a0.x, a0.y);
            A.p32[1] = pack_bf2(a0.z, a0.w);
            A.p32[2] = pack_bf2(a1.x, a1.y);
            A.p32[3] = pack_bf2(a1.z, a1.w);
            A.p32[4] = pack_bf2(a2.x, a2.y);
            A.p32[5] = pack_bf2(a2.z, a2.w);
            A.p32[6] = pack_bf2(a3.x, a3.y);
            A.p32[7] = pack_bf2(a3.z, a3.w);
        }

        // ---- B fragments straight from L2-resident bf16 W_proj (2x 16B each)
        Frag16 B0, B1;
        {
            const unsigned short* wp0 = Wb + (size_t)j0 * DD + kc * 32 + k0;
            B0.q[0] = *(const uint4*)wp0;
        B0.q[1] = *(const uint4*)(wp0 + 16);
            const unsigned short* wp1 = Wb + (size_t)(j0 + 16) * DD + kc * 32 + k0;
            B1.q[0] = *(const uint4*)wp1;
        B1.q[1] = *(const uint4*)(wp1 + 16);
        }

        acc0 = __builtin_amdgcn_wmma_f32_16x16x32_bf16(
                   false, A.v, false, B0.v, (short)0, acc0, false, false);
        acc1 = __builtin_amdgcn_wmma_f32_16x16x32_bf16(
                   false, A.v, false, B1.v, (short)0, acc1, false, false);

        __syncthreads();   // before the next issue() overwrites this buffer
    }

    // ---- store H tiles (C layout: M = 8*(lane/16)+i, N = lane%16)
    {
        const int mofs = (lane < 16) ? 0 : 8;
        #pragma unroll
        for (int i = 0; i < 8; ++i) {
            const int row = l0 + wid * 16 + mofs + i;
            unsigned short* hp = H + ((size_t)n * LL + row) * DH;
            hp[j0]      = f2bf1(acc0[i]);
            hp[16 + j0] = f2bf1(acc1[i]);
        }
    }

    // ---- first_order reduction: sum_l w_seq[l] * (x_l . w_token)
    red[tid] = w_seq[l0 + rl] * s_fo;
    __syncthreads();
    for (int s = 128; s > 0; s >>= 1) {
        if (tid < s) red[tid] += red[tid + s];
        __syncthreads();
    }
    if (tid == 0) atomicAdd(&out[n], red[0]);
}

// ---------------------------------------------------------------------------
// Kernel 2: second_order. One 16x16 tile of (H H^T)*scale per wave; only
// tile pairs tm >= tl, strict row<col mask. H (2 MB) stays L2-resident;
// priors (67 MB) streams non-temporally. grid: JOBS/8 blocks, 256 threads.
// ---------------------------------------------------------------------------
__global__ void attn_kernel(const unsigned short* __restrict__ H,
                            const float* __restrict__ priors,
                            const float* __restrict__ interaction_scale,
                            float* __restrict__ out) {
    const int tid  = threadIdx.x;
    const int lane = tid & 31;
    const int wid  = tid >> 5;

    const int job = blockIdx.x * 8 + wid;      // < JOBS by construction
    const int n   = job / TRI;
    int p         = job % TRI;

    int tl = 0;
    while (p >= (LL / 16) - tl) { p -= (LL / 16) - tl; ++tl; }
    const int tm = tl + p;

    // A: 16 H rows at l-tile (per-lane: 8 halfs at K=kb, 8 at K=kb+16)
    Frag16 A, B;
    {
        const unsigned short* hp =
            H + ((size_t)n * LL + tl * 16 + (lane & 15)) * DH + ((lane < 16) ? 0 : 8);
        A.q[0] = *(const uint4*)hp;
        A.q[1] = *(const uint4*)(hp + 16);
    }
    // B: 16 H rows at m-tile, 8 halfs at K=kb, 8 at K=kb+16 per lane (row-contiguous
    // in H == the K x 16 B layout of H^T, no transpose needed)
    {
        const unsigned short* hq =
            H + ((size_t)n * LL + tm * 16 + (lane & 15)) * DH + ((lane < 16) ? 0 : 8);
        B.q[0] = *(const uint4*)hq;
        B.q[1] = *(const uint4*)(hq + 16);
    }

    v8f acc = {};
    acc = __builtin_amdgcn_wmma_f32_16x16x32_bf16(
              false, A.v, false, B.v, (short)0, acc, false, false);

    // priors: single-use 67 MB stream -> unconditional non-temporal loads
    // (always in-range), masked by select so the 8 loads form one clause
    const float* P = priors + (size_t)n * LL * LL;
    const int col  = tm * 16 + (lane & 15);
    const int mofs = (lane < 16) ? 0 : 8;
    float s = 0.0f;
    #pragma unroll
    for (int i = 0; i < 8; ++i) {
        const int row = tl * 16 + mofs + i;
        const float pv = __builtin_nontemporal_load(&P[(size_t)row * LL + col]);
        s += (row < col) ? acc[i] * pv : 0.0f;
    }
    s *= SCALE_ATTN * interaction_scale[0];

    #pragma unroll
    for (int o = 16; o > 0; o >>= 1)
        s += __shfl_xor(s, o, 32);
    if (lane == 0) atomicAdd(&out[n], s);
}

// ---------------------------------------------------------------------------
// Launch
// ---------------------------------------------------------------------------
extern "C" void kernel_launch(void* const* d_in, const int* in_sizes, int n_in,
                              void* d_out, int out_size, void* d_ws, size_t ws_size,
                              hipStream_t stream) {
    const float* x      = (const float*)d_in[0];
    const float* priors = (const float*)d_in[1];
    const float* w_tok  = (const float*)d_in[2];
    const float* w_seq  = (const float*)d_in[3];
    const float* b_seq  = (const float*)d_in[4];
    const float* W_proj = (const float*)d_in[5];
    const float* iscale = (const float*)d_in[6];
    float* out = (float*)d_out;

    // workspace: H bf16 (2 MB) then W_proj bf16 (80 KB)
    unsigned short* H  = (unsigned short*)d_ws;
    unsigned short* Wb = (unsigned short*)((char*)d_ws + (size_t)NB * LL * DH * 2);

    init_out_kernel<<<1, 64, 0, stream>>>(out, b_seq);
    wconv_kernel<<<(DH * DD / 2) / 256, 256, 0, stream>>>(W_proj, (unsigned*)Wb);
    proj_kernel<<<NB * (LL / ROWS), 256, 0, stream>>>(x, w_tok, w_seq, Wb, H, out);
    attn_kernel<<<JOBS / 8, 256, 0, stream>>>(H, priors, iscale, out);
}
